// Retriever_44358422233672
// MI455X (gfx1250) — matrix operand order repaired and checked by reading, not compiled, not run on hardware
//
#include <hip/hip_runtime.h>
#include <hip/hip_bf16.h>

// CDNA5 / gfx1250 wave32 WMMA + TDM path
typedef __attribute__((ext_vector_type(16))) __bf16 v16bf;
typedef __attribute__((ext_vector_type(8)))  float  v8f;
typedef __attribute__((ext_vector_type(4)))  unsigned int u32x4;
typedef __attribute__((ext_vector_type(8)))  int i32x8;
typedef __attribute__((ext_vector_type(4)))  int i32x4;

struct __align__(16) U4 { unsigned int x, y, z, w; };
struct __align__(16) F4 { float x, y, z, w; };
struct __align__(8)  U2 { unsigned int x, y; };

union FragBF { U4 q[2]; v16bf v; };

#if defined(__has_builtin)
#if __has_builtin(__builtin_amdgcn_tensor_load_to_lds) && __has_builtin(__builtin_amdgcn_s_wait_tensorcnt)
#define USE_TDM 1
#endif
#endif
#ifndef USE_TDM
#define USE_TDM 0
#endif

#if __has_include(<hip/amd_detail/amd_gfx1250_TDM.h>)
#define TDM_SIX_ARG 1   // amdgpu-toolchain / therock headers -> 6-arg builtin
#else
#define TDM_SIX_ARG 0   // ROCm 7.2 -> 5-arg builtin
#endif

__device__ __forceinline__ unsigned short f2bf(float f) {
  unsigned int u = __builtin_bit_cast(unsigned int, f);
  unsigned int r = (u + 0x7FFFu + ((u >> 16) & 1u)) >> 16;  // round-to-nearest-even
  return (unsigned short)r;
}

#if USE_TDM
// Issue a TDM DMA: rows x 32 bf16 tile (row stride = K elements) -> LDS.
// D# layout per CDNA5 ISA 8.3/8.4: group0 = {count/type, lds_addr, global_addr},
// group1 = {data_size, tensor dims, tile dims, dim0 stride}.
__device__ __forceinline__ void tdm_load_tile(unsigned lds_off,
                                              const unsigned short* gptr,
                                              unsigned K, unsigned rows) {
  unsigned long long ga = (unsigned long long)(uintptr_t)gptr;
  u32x4 g0;
  g0[0] = 1u;                                                  // count=1 (valid), user mode
  g0[1] = lds_off;                                             // lds_addr (bytes)
  g0[2] = (unsigned)ga;                                        // global_addr[31:0]
  g0[3] = ((unsigned)(ga >> 32) & 0x01FFFFFFu) | (2u << 30);   // global_addr[56:32], type=2
  i32x8 g1;
  g1[0] = (int)0x00010000u;                                    // data_size=1 -> 2 bytes
  g1[1] = (int)((K & 0xFFFFu) << 16);                          // tensor_dim0[15:0]
  g1[2] = (int)((K >> 16) | ((rows & 0xFFFFu) << 16));         // dim0[31:16] | dim1[15:0]
  g1[3] = (int)((rows >> 16) | (32u << 16));                   // dim1[31:16] | tile_dim0=32
  g1[4] = (int)rows;                                           // tile_dim1=rows, tile_dim2=0
  g1[5] = (int)K;                                              // tensor_dim0_stride[31:0]
  g1[6] = 0;                                                   // stride[47:32]=0
  g1[7] = 0;
  i32x4 z4 = {};
#if TDM_SIX_ARG
  i32x8 z8 = {};
  __builtin_amdgcn_tensor_load_to_lds(g0, g1, z4, z4, z8, 0);
#else
  __builtin_amdgcn_tensor_load_to_lds(g0, g1, z4, z4, 0);
#endif
}
#endif

// ---------- f32 -> bf16 weight conversion (n multiple of 4) ----------
__global__ __launch_bounds__(256)
void cvt_bf16_kernel(const float* __restrict__ s, unsigned short* __restrict__ d, int n) {
  int i = (blockIdx.x * 256 + threadIdx.x) * 4;
  if (i >= n) return;
  F4 v = *(const F4*)(s + i);
  U2 p;
  p.x = (unsigned)f2bf(v.x) | ((unsigned)f2bf(v.y) << 16);
  p.y = (unsigned)f2bf(v.z) | ((unsigned)f2bf(v.w) << 16);
  *(U2*)(d + i) = p;
}

// ---------- per-row label aggregation + X = [bf16(cls) | bf16(agg)] ----------
// This kernel carries ~128 MB (sims + labels) => the roofline term; b128 loads.
__global__ __launch_bounds__(256)
void build_x_kernel(const float* __restrict__ features,
                    const float* __restrict__ sims,
                    const int* __restrict__ labels,
                    unsigned short* __restrict__ X) {
  const int b = blockIdx.x, t = threadIdx.x;
  __shared__ float bins[128];
  if (t < 128) bins[t] = 0.0f;
  __syncthreads();
  const F4* s4 = (const F4*)(sims + (size_t)b * 16384);
  const U4* l4 = (const U4*)(labels + (size_t)b * 16384);
#pragma unroll 4
  for (int n = t; n < 4096; n += 256) {     // 16 iters x (2 b128 loads + 4 ds_add_f32)
    F4 sv = s4[n];
    U4 lv = l4[n];
    atomicAdd(&bins[lv.x & 127], sv.x);
    atomicAdd(&bins[lv.y & 127], sv.y);
    atomicAdd(&bins[lv.z & 127], sv.z);
    atomicAdd(&bins[lv.w & 127], sv.w);
  }
  __syncthreads();
  unsigned short* xr = X + (size_t)b * 1152;
  F4 v = ((const F4*)(features + (size_t)b * 16 * 1024))[t];   // CLS row
  U2 p;
  p.x = (unsigned)f2bf(v.x) | ((unsigned)f2bf(v.y) << 16);
  p.y = (unsigned)f2bf(v.z) | ((unsigned)f2bf(v.w) << 16);
  *(U2*)(xr + t * 4) = p;
  if (t < 128) xr[1024 + t] = f2bf(bins[t] * (1.0f / 16384.0f));
}

// ---------- C = act(A(MxK,bf16) * B(NxK,bf16)^T + bias) ----------
// MODE 0: tanh, store bf16.  MODE 1: identity, store f32.
// Block tile 128x64, BK=32, 256 threads = 8 waves, wave tile 32x32.
// Tiles are fed by the Tensor Data Mover (double-buffered) when available.
template<int MODE>
__global__ __launch_bounds__(256)
void gemm_bt_kernel(const unsigned short* __restrict__ A,
                    const unsigned short* __restrict__ Bm,
                    const float* __restrict__ bias,
                    void* __restrict__ Cout,
                    int M, int N, int K) {
#if USE_TDM
  __shared__ U4 sA4[2][512];  // 2 x 8 KB
  __shared__ U4 sB4[2][256];  // 2 x 4 KB
#else
  __shared__ U4 sA4[1][512];
  __shared__ U4 sB4[1][256];
#endif
  const int t    = threadIdx.x;
  const int lane = t & 31;
  const int wave = t >> 5;
  const int r = lane & 15;          // row/col within 16x16 tile
  const int h = lane >> 4;          // half-wave select
  const int wm = (wave >> 1) * 32;  // wave M offset in block tile
  const int wn = (wave & 1) * 32;   // wave N offset in block tile
  const int bm = blockIdx.y * 128;
  const int bn = blockIdx.x * 64;

  v8f acc[2][2] = {};
  const unsigned short* Ag = A  + (size_t)bm * K;
  const unsigned short* Bg = Bm + (size_t)bn * K;

#if USE_TDM
  const unsigned ldsA[2] = { (unsigned)(uintptr_t)&sA4[0][0],
                             (unsigned)(uintptr_t)&sA4[1][0] };
  const unsigned ldsB[2] = { (unsigned)(uintptr_t)&sB4[0][0],
                             (unsigned)(uintptr_t)&sB4[1][0] };
  if (t < 32) {                       // wave 0 drives the TDM
    tdm_load_tile(ldsA[0], Ag, (unsigned)K, 128u);
    tdm_load_tile(ldsB[0], Bg, (unsigned)K, 64u);
  }
  int cur = 0;
#endif

  for (int k0 = 0; k0 < K; k0 += 32) {
#if USE_TDM
    if (t < 32) {
      if (k0 + 32 < K) {              // prefetch next tile pair into other buffer
        tdm_load_tile(ldsA[cur ^ 1], Ag + k0 + 32, (unsigned)K, 128u);
        tdm_load_tile(ldsB[cur ^ 1], Bg + k0 + 32, (unsigned)K, 64u);
        __builtin_amdgcn_s_wait_tensorcnt(2);   // current pair landed (in-order)
      } else {
        __builtin_amdgcn_s_wait_tensorcnt(0);
      }
    }
    __syncthreads();
    const U4* a_base = &sA4[cur][0];
    const U4* b_base = &sB4[cur][0];
#else
    {
      int row = t >> 2, c = t & 3;
      sA4[0][t]       = *(const U4*)(Ag + (size_t)row * K + k0 + c * 8);
      int i1 = t + 256;
      int row1 = i1 >> 2, c1 = i1 & 3;
      sA4[0][i1]      = *(const U4*)(Ag + (size_t)row1 * K + k0 + c1 * 8);
      sB4[0][t]       = *(const U4*)(Bg + (size_t)row * K + k0 + c * 8);
    }
    __syncthreads();
    const U4* a_base = &sA4[0][0];
    const U4* b_base = &sB4[0][0];
#endif

    FragBF a[2], b[2];
#pragma unroll
    for (int ti = 0; ti < 2; ++ti) {
      // ISA 16-bit A 16x32 layout: lanes0-15 K0-7/K16-23, lanes16-31 K8-15/K24-31
      const U4* rp = a_base + (wm + 16 * ti + r) * 4;
      a[ti].q[0] = rp[h];
      a[ti].q[1] = rp[2 + h];
    }
#pragma unroll
    for (int tj = 0; tj < 2; ++tj) {
      // B 32x16 layout: lanes0-15 K0-7/K16-23, lanes16-31 K8-15/K24-31 (col = r)
      const U4* rp = b_base + (wn + 16 * tj + r) * 4;
      b[tj].q[0] = rp[h];
      b[tj].q[1] = rp[2 + h];
    }
#pragma unroll
    for (int ti = 0; ti < 2; ++ti)
#pragma unroll
      for (int tj = 0; tj < 2; ++tj)
        acc[ti][tj] = __builtin_amdgcn_wmma_f32_16x16x32_bf16(
            false, a[ti].v, false, b[tj].v, (short)0, acc[ti][tj], false, false);
    __syncthreads();
#if USE_TDM
    cur ^= 1;
#endif
  }

  // epilogue: D tile layout — vgpr j holds (m = 8*h + j, n = r)
#pragma unroll
  for (int ti = 0; ti < 2; ++ti) {
#pragma unroll
    for (int tj = 0; tj < 2; ++tj) {
      int gn = bn + wn + 16 * tj + r;
      float bv = bias[gn];
      int gm0 = bm + wm + 16 * ti + 8 * h;
#pragma unroll
      for (int j = 0; j < 8; ++j) {
        float v = acc[ti][tj][j] + bv;
        if (MODE == 0)
          ((unsigned short*)Cout)[(size_t)(gm0 + j) * N + gn] = f2bf(tanhf(v));
        else
          ((float*)Cout)[(size_t)(gm0 + j) * N + gn] = v;
      }
    }
  }
}

extern "C" void kernel_launch(void* const* d_in, const int* in_sizes, int n_in,
                              void* d_out, int out_size, void* d_ws, size_t ws_size,
                              hipStream_t stream) {
  const float* features = (const float*)d_in[0];  // (1024,16,1024)
  const float* sims     = (const float*)d_in[1];  // (1024,16384)
  const int*   labels   = (const int*)d_in[2];    // (1024,16384)
  const float* dense_w  = (const float*)d_in[3];  // (1024,1152)
  const float* dense_b  = (const float*)d_in[4];  // (1024,)
  const float* out_w    = (const float*)d_in[5];  // (128,1024)
  const float* out_b    = (const float*)d_in[6];  // (128,)

  char* ws = (char*)d_ws;
  unsigned short* X  = (unsigned short*)(ws);              // 1024*1152 bf16
  unsigned short* Wd = (unsigned short*)(ws + 2359296);    // 1024*1152 bf16
  unsigned short* Y  = (unsigned short*)(ws + 4718592);    // 1024*1024 bf16
  unsigned short* Wo = (unsigned short*)(ws + 6815744);    // 128*1024  bf16

  cvt_bf16_kernel<<<(1024 * 1152) / 1024, 256, 0, stream>>>(dense_w, Wd, 1024 * 1152);
  cvt_bf16_kernel<<<(128 * 1024) / 1024, 256, 0, stream>>>(out_w, Wo, 128 * 1024);
  build_x_kernel<<<1024, 256, 0, stream>>>(features, sims, labels, X);

  // y = tanh(X @ Wd^T + dense_b)  : M=1024 N=1024 K=1152
  gemm_bt_kernel<0><<<dim3(1024 / 64, 1024 / 128), 256, 0, stream>>>(
      X, Wd, dense_b, (void*)Y, 1024, 1024, 1152);
  // out = Y @ Wo^T + out_b        : M=1024 N=128  K=1024
  gemm_bt_kernel<1><<<dim3(128 / 64, 1024 / 128), 256, 0, stream>>>(
      Y, Wo, out_b, d_out, 1024, 128, 1024);
}
